// VIT_23587960389956
// MI455X (gfx1250) — matrix operand order repaired and checked by reading, not compiled, not run on hardware
//
#include <hip/hip_runtime.h>
#include <hip/hip_bf16.h>
#include <cstdint>

// ---------------------------------------------------------------------------
// Video-ViT forward for MI455X (gfx1250).  All GEMMs (patch-embed, QKV, attn
// scores, attn*V, out-proj, MLP, head) run on v_wmma_f32_16x16x32_bf16 with
// fp32 accumulation.  Attention is fused flash-style (never materializes NxN).
// CDNA5 paths used: v_wmma, global_load_async_to_lds_b128 (ASYNCcnt),
// DPP16 cross-lane reductions, s_wait_* split counters.
// ---------------------------------------------------------------------------

typedef __bf16 bf16;
typedef __bf16 v16bf __attribute__((ext_vector_type(16)));
typedef float  v8f   __attribute__((ext_vector_type(8)));

#define EMBED   768
#define HEADS   8
#define HDIM    96
#define DEPTH   12
#define MLPH    3072
#define NPATCH  1568
#define NCLS    1000
#define BATCH   2
#define MROWS   (BATCH * NPATCH)   /* 3136 */
#define EPSLN   1e-5f

union Frag {
  v16bf v;
  uint4 q[2];
};

__device__ __forceinline__ bf16 f2bf(float f) { return (bf16)f; }

__device__ __forceinline__ v8f vzero8() {
  v8f z;
#pragma unroll
  for (int i = 0; i < 8; ++i) z[i] = 0.0f;
  return z;
}

#define WMMA_BF16(A, B, C)                                                    \
  __builtin_amdgcn_wmma_f32_16x16x32_bf16(false, (A), false, (B), (short)0,   \
                                          (C), false, false)

// --- DPP16 cross-lane reduction within each 16-lane row (no LDS traffic) ---
template <int CTRL>
__device__ __forceinline__ float dppmove(float x) {
  return __int_as_float(
      __builtin_amdgcn_update_dpp(0, __float_as_int(x), CTRL, 0xF, 0xF, true));
}
__device__ __forceinline__ float red16_max(float x) {
  x = fmaxf(x, dppmove<0xB1>(x));   // quad_perm [1,0,3,2]  (xor 1)
  x = fmaxf(x, dppmove<0x4E>(x));   // quad_perm [2,3,0,1]  (xor 2)
  x = fmaxf(x, dppmove<0x141>(x));  // row_half_mirror      (xor 7)
  x = fmaxf(x, dppmove<0x140>(x));  // row_mirror           (xor 15)
  return x;
}
__device__ __forceinline__ float red16_sum(float x) {
  x += dppmove<0xB1>(x);
  x += dppmove<0x4E>(x);
  x += dppmove<0x141>(x);
  x += dppmove<0x140>(x);
  return x;
}

// Async copy of 64 contiguous bytes global -> LDS (4x b128, ASYNCcnt path).
// INST_OFFSET is applied to both the LDS and the global address (ISA 10.x).
__device__ __forceinline__ void async_copy64(unsigned lds_off,
                                             const bf16* gptr) {
  asm volatile(
      "global_load_async_to_lds_b128 %0, %1, off\n\t"
      "global_load_async_to_lds_b128 %0, %1, off offset:16\n\t"
      "global_load_async_to_lds_b128 %0, %1, off offset:32\n\t"
      "global_load_async_to_lds_b128 %0, %1, off offset:48"
      :: "v"(lds_off), "v"(gptr)
      : "memory");
}

// ---------------------------------------------------------------------------
// Generic tiled GEMM:  out[M,Nout] = A[M,K](bf16) * W[Nout,K](f32->bf16)^T
// Epilogue modes:
//   0: fp32 = acc + bias                       (head)
//   1: fp32 = acc + bias + residual            (out-proj / fc2 into h)
//   2: bf16 = acc                              (QKV, no bias)
//   3: bf16 = gelu_exact(acc + bias)           (fc1)
//   4: fp32 = acc + bias + pos_encoding(n,c)   (patch embed)
// ---------------------------------------------------------------------------
template <int EPI>
__global__ __launch_bounds__(256) void gemm_bf16(
    const bf16* __restrict__ A, const float* __restrict__ W, int M, int Nout,
    int K, const float* __restrict__ bias, const float* res, float* outF,
    bf16* outB) {
  __shared__ __align__(16) bf16 As[128 * 72];
  __shared__ __align__(16) bf16 Ws[128 * 72];

  const int tid  = threadIdx.x;
  const int lane = tid & 31;
  const int wid  = tid >> 5;
  const int l15  = lane & 15;
  const int half = lane >> 4;
  const int wrow = wid & 3;   // M sub-block *32
  const int wcol = wid >> 2;  // N sub-block *64
  const int tileM = blockIdx.y * 128;
  const int tileN = blockIdx.x * 128;

  v8f acc[2][4];
#pragma unroll
  for (int i = 0; i < 2; ++i)
#pragma unroll
    for (int j = 0; j < 4; ++j) acc[i][j] = vzero8();

  const int ldRow = tid >> 1;  // 0..127  (16 rows per wave -> guard is wave-uniform)
  const int ldSeg = tid & 1;   // column segment *32
  const unsigned ldsA =
      (unsigned)(uintptr_t)&As[ldRow * 72 + ldSeg * 32];  // LDS byte offset

  for (int k0 = 0; k0 < K; k0 += 64) {
    __syncthreads();  // previous tile fully consumed by all waves
    {  // stage A tile via async DMA (raw bf16 bytes, no conversion needed)
      int ar = tileM + ldRow;
      if (ar < M) {
        async_copy64(ldsA, A + (size_t)ar * K + k0 + ldSeg * 32);
      } else {
        uint4 z; z.x = z.y = z.z = z.w = 0u;
        uint4* dst = (uint4*)&As[ldRow * 72 + ldSeg * 32];
        dst[0] = z; dst[1] = z; dst[2] = z; dst[3] = z;
      }
    }
    {  // stage W tile through VGPRs, converting f32 -> bf16 on the fly
      int wr = tileN + ldRow;
      bf16 tmp[32];
      if (wr < Nout) {
        const float* wrow_p = W + (size_t)wr * K + k0 + ldSeg * 32;
        if (k0 + 64 < K) __builtin_prefetch(wrow_p + 64, 0, 0);
        const float4* wp = (const float4*)wrow_p;
#pragma unroll
        for (int i = 0; i < 8; ++i) {
          float4 f = wp[i];
          tmp[4 * i + 0] = f2bf(f.x);
          tmp[4 * i + 1] = f2bf(f.y);
          tmp[4 * i + 2] = f2bf(f.z);
          tmp[4 * i + 3] = f2bf(f.w);
        }
      } else {
#pragma unroll
        for (int i = 0; i < 32; ++i) tmp[i] = (bf16)0.0f;
      }
#pragma unroll
      for (int i = 0; i < 32; ++i) Ws[ldRow * 72 + ldSeg * 32 + i] = tmp[i];
    }
    asm volatile("s_wait_asynccnt 0x0" ::: "memory");  // A-tile DMA done
    __syncthreads();                                   // visible to all waves

#pragma unroll
    for (int kk = 0; kk < 64; kk += 32) {
      Frag fa[2], fw[4];
#pragma unroll
      for (int si = 0; si < 2; ++si) {
        const bf16* p = &As[(wrow * 32 + si * 16 + l15) * 72 + kk];
        fa[si].q[0] = *(const uint4*)(p + half * 16);
        fa[si].q[1] = *(const uint4*)(p + half * 16 + 8);
      }
#pragma unroll
      for (int sj = 0; sj < 4; ++sj) {
        const bf16* p = &Ws[(wcol * 64 + sj * 16 + l15) * 72 + kk + half * 16];
        fw[sj].q[0] = *(const uint4*)(p);
        fw[sj].q[1] = *(const uint4*)(p + 8);
      }
#pragma unroll
      for (int si = 0; si < 2; ++si)
#pragma unroll
        for (int sj = 0; sj < 4; ++sj)
          acc[si][sj] = WMMA_BF16(fa[si].v, fw[sj].v, acc[si][sj]);
    }
  }

  // epilogue — C layout: element (m = r + 8*half, n = l15)
#pragma unroll
  for (int si = 0; si < 2; ++si) {
#pragma unroll
    for (int sj = 0; sj < 4; ++sj) {
      int gn = tileN + wcol * 64 + sj * 16 + l15;
      if (gn >= Nout) continue;
      int gmBase = tileM + wrow * 32 + si * 16 + 8 * half;
#pragma unroll
      for (int r = 0; r < 8; ++r) {
        int gm = gmBase + r;
        if (gm >= M) continue;
        float a = acc[si][sj][r];
        size_t idx = (size_t)gm * Nout + gn;
        if (EPI == 0) {
          outF[idx] = a + bias[gn];
        } else if (EPI == 1) {
          outF[idx] = a + bias[gn] + res[idx];
        } else if (EPI == 2) {
          outB[idx] = f2bf(a);
        } else if (EPI == 3) {
          float xg = a + bias[gn];
          outB[idx] = f2bf(0.5f * xg * (1.0f + erff(xg * 0.70710678118f)));
        } else if (EPI == 4) {
          int p = gm % NPATCH;
          float ex = (float)(2 * (gn >> 1)) * (1.0f / 768.0f);
          float ang = (float)p * __powf(10000.0f, -ex);
          float pos = (gn & 1) ? __cosf(ang) : __sinf(ang);
          outF[idx] = a + bias[gn] + pos;
        }
      }
    }
  }
}

// ---------------------------------------------------------------------------
// Flash attention: 1 wave per (b, h, 16-query tile); 49 key tiles of 32.
// qkv: [B*N, 2304] bf16  (q | k | v, head-major, d contiguous)
// vt:  [B, H, 96, N] bf16 (d-major transposed V)
// o:   [B*N, 768] bf16
// ---------------------------------------------------------------------------
__global__ __launch_bounds__(128) void attn_kernel(const bf16* __restrict__ qkv,
                                                   const bf16* __restrict__ vt,
                                                   bf16* __restrict__ o) {
  __shared__ __align__(16) bf16 plds[4][16 * 32];
  const int tid  = threadIdx.x;
  const int lane = tid & 31;
  const int wid  = tid >> 5;
  const int l15  = lane & 15;
  const int half = lane >> 4;

  const int bid = blockIdx.x;
  const int qt  = (bid % 25) * 4 + wid;  // query tile
  const int bh  = bid / 25;
  const int b   = bh >> 3;
  const int h   = bh & 7;
  if (qt >= 98) return;  // uniform per-wave exit; no block barriers used

  // Q fragments: A-layout, 3 chunks over d (96 = 3*32)
  Frag fq[3];
#pragma unroll
  for (int kc = 0; kc < 3; ++kc) {
    const bf16* qb =
        qkv + ((size_t)(b * NPATCH + qt * 16 + l15)) * 2304 + h * 96 + kc * 32;
    fq[kc].q[0] = *(const uint4*)(qb + half * 16);
    fq[kc].q[1] = *(const uint4*)(qb + half * 16 + 8);
  }

  v8f ov[6];
#pragma unroll
  for (int i = 0; i < 6; ++i) ov[i] = vzero8();
  float mrow[8], lrow[8];
#pragma unroll
  for (int r = 0; r < 8; ++r) { mrow[r] = -3.0e38f; lrow[r] = 0.0f; }

  for (int kt = 0; kt < 49; ++kt) {
    v8f s0 = vzero8(), s1 = vzero8();
#pragma unroll
    for (int kc = 0; kc < 3; ++kc) {
      Frag fk0, fk1;
      const bf16* kb0 = qkv + ((size_t)(b * NPATCH + kt * 32 + l15)) * 2304 +
                        768 + h * 96 + kc * 32 + half * 16;
      const bf16* kb1 = kb0 + (size_t)16 * 2304;
      fk0.q[0] = *(const uint4*)(kb0);
      fk0.q[1] = *(const uint4*)(kb0 + 8);
      fk1.q[0] = *(const uint4*)(kb1);
      fk1.q[1] = *(const uint4*)(kb1 + 8);
      s0 = WMMA_BF16(fq[kc].v, fk0.v, s0);
      s1 = WMMA_BF16(fq[kc].v, fk1.v, s1);
    }
    // online softmax per row; DPP16 reductions stay inside each 16-lane half
#pragma unroll
    for (int r = 0; r < 8; ++r) {
      float v0 = s0[r], v1 = s1[r];
      float mr   = red16_max(fmaxf(v0, v1));
      float mn   = fmaxf(mrow[r], mr);
      float corr = __expf(mrow[r] - mn);
      float p0   = __expf(v0 - mn);
      float p1   = __expf(v1 - mn);
      float rs   = red16_sum(p0 + p1);
      lrow[r] = lrow[r] * corr + rs;
      mrow[r] = mn;
#pragma unroll
      for (int dc = 0; dc < 6; ++dc) ov[dc][r] *= corr;
      int m = r + 8 * half;
      plds[wid][m * 32 + l15]      = f2bf(p0);
      plds[wid][m * 32 + 16 + l15] = f2bf(p1);
    }
    // cross-lane LDS visibility within the wave (DS ops are in-order per wave)
    asm volatile("s_wait_dscnt 0x0" ::: "memory");
    Frag fp;
    {
      const bf16* pb = &plds[wid][l15 * 32];
      fp.q[0] = *(const uint4*)(pb + half * 16);
      fp.q[1] = *(const uint4*)(pb + half * 16 + 8);
    }
#pragma unroll
    for (int dc = 0; dc < 6; ++dc) {
      Frag fv;
      const bf16* vb = vt +
                       ((size_t)((b * 8 + h) * 96 + dc * 16 + l15)) * NPATCH +
                       kt * 32 + half * 16;
      fv.q[0] = *(const uint4*)(vb);
      fv.q[1] = *(const uint4*)(vb + 8);
      ov[dc] = WMMA_BF16(fp.v, fv.v, ov[dc]);
    }
  }

#pragma unroll
  for (int dc = 0; dc < 6; ++dc) {
#pragma unroll
    for (int r = 0; r < 8; ++r) {
      int row = qt * 16 + r + 8 * half;
      o[((size_t)(b * NPATCH + row)) * 768 + h * 96 + dc * 16 + l15] =
          f2bf(ov[dc][r] / lrow[r]);
    }
  }
}

// ---------------------------------------------------------------------------
// LayerNorm over C=768: fp32 in -> bf16 out.  One block per row.
// ---------------------------------------------------------------------------
__global__ __launch_bounds__(256) void ln_kernel(const float* __restrict__ x,
                                                 const float* __restrict__ scale,
                                                 const float* __restrict__ bias,
                                                 bf16* __restrict__ y) {
  __shared__ float r1[256], r2[256];
  const int row = blockIdx.x;
  const int tid = threadIdx.x;
  const float* xr = x + (size_t)row * 768;
  float v0 = xr[tid], v1 = xr[tid + 256], v2 = xr[tid + 512];
  r1[tid] = v0 + v1 + v2;
  r2[tid] = v0 * v0 + v1 * v1 + v2 * v2;
  __syncthreads();
  for (int s = 128; s > 0; s >>= 1) {
    if (tid < s) { r1[tid] += r1[tid + s]; r2[tid] += r2[tid + s]; }
    __syncthreads();
  }
  float mean = r1[0] * (1.0f / 768.0f);
  float var  = r2[0] * (1.0f / 768.0f) - mean * mean;
  float rstd = rsqrtf(var + EPSLN);
  bf16* yr = y + (size_t)row * 768;
  yr[tid]       = f2bf((v0 - mean) * rstd * scale[tid]       + bias[tid]);
  yr[tid + 256] = f2bf((v1 - mean) * rstd * scale[tid + 256] + bias[tid + 256]);
  yr[tid + 512] = f2bf((v2 - mean) * rstd * scale[tid + 512] + bias[tid + 512]);
}

// ---------------------------------------------------------------------------
// im2col for Conv3d(kernel==stride==(2,16,16)): x[B,3,16,224,224] -> P[3136,1536]
// ---------------------------------------------------------------------------
__global__ __launch_bounds__(256) void im2col_kernel(const float* __restrict__ x,
                                                     bf16* __restrict__ P) {
  const int m  = blockIdx.x;
  const int b  = m / NPATCH;
  const int n  = m % NPATCH;
  const int t  = n / 196;
  const int py = (n % 196) / 14;
  const int px = n % 14;
  bf16* pr = P + (size_t)m * 1536;
  for (int k = threadIdx.x; k < 1536; k += 256) {
    int ci = k >> 9;
    int r  = k & 511;
    int kd = r >> 8;
    int r2 = r & 255;
    int kh = r2 >> 4;
    int kw = r2 & 15;
    size_t xi = ((((size_t)b * 3 + ci) * 16 + (2 * t + kd)) * 224 +
                 (16 * py + kh)) * 224 + (16 * px + kw);
    pr[k] = f2bf(x[xi]);
  }
}

// ---------------------------------------------------------------------------
// V transpose: qkv V slice -> vt[B,H,96,N] (d-major so PV B-frags are contiguous)
// ---------------------------------------------------------------------------
__global__ __launch_bounds__(256) void vtrans_kernel(const bf16* __restrict__ qkv,
                                                     bf16* __restrict__ vt) {
  const int row = blockIdx.x;  // (b*8+h)*96 + d, 1536 rows
  const int d   = row % 96;
  const int bh  = row / 96;
  const int h   = bh & 7;
  const int b   = bh >> 3;
  const bf16* src = qkv + (size_t)b * NPATCH * 2304 + 1536 + h * 96 + d;
  bf16* dst = vt + (size_t)row * NPATCH;
  for (int n = threadIdx.x; n < NPATCH; n += 256)
    dst[n] = src[(size_t)n * 2304];
}

// ---------------------------------------------------------------------------
// Mean pool over N + final LayerNorm -> bf16 [B,768]
// ---------------------------------------------------------------------------
__global__ __launch_bounds__(768) void pool_ln_kernel(
    const float* __restrict__ h, const float* __restrict__ scale,
    const float* __restrict__ bias, bf16* __restrict__ out) {
  __shared__ float r1[768], r2[768];
  const int b = blockIdx.x, c = threadIdx.x;
  float s = 0.0f;
  const float* hp = h + (size_t)b * NPATCH * 768 + c;
  for (int n = 0; n < NPATCH; ++n) s += hp[(size_t)n * 768];
  float mc = s * (1.0f / (float)NPATCH);
  r1[c] = mc;
  r2[c] = mc * mc;
  __syncthreads();
  for (int st = 512; st > 0; st >>= 1) {
    if (c < st && c + st < 768) { r1[c] += r1[c + st]; r2[c] += r2[c + st]; }
    __syncthreads();
  }
  float mean = r1[0] * (1.0f / 768.0f);
  float var  = r2[0] * (1.0f / 768.0f) - mean * mean;
  float rstd = rsqrtf(var + EPSLN);
  out[b * 768 + c] = f2bf((mc - mean) * rstd * scale[c] + bias[c]);
}

// ---------------------------------------------------------------------------
extern "C" void kernel_launch(void* const* d_in, const int* in_sizes, int n_in,
                              void* d_out, int out_size, void* d_ws,
                              size_t ws_size, hipStream_t stream) {
  (void)in_sizes; (void)n_in; (void)out_size; (void)ws_size;
  const float* x      = (const float*)d_in[0];
  const float* conv_w = (const float*)d_in[1];
  const float* conv_b = (const float*)d_in[2];
  const float* n1s    = (const float*)d_in[3];
  const float* n1b    = (const float*)d_in[4];
  const float* qkv_w  = (const float*)d_in[5];
  const float* out_w  = (const float*)d_in[6];
  const float* out_b  = (const float*)d_in[7];
  const float* n2s    = (const float*)d_in[8];
  const float* n2b    = (const float*)d_in[9];
  const float* fc1_w  = (const float*)d_in[10];
  const float* fc1_b  = (const float*)d_in[11];
  const float* fc2_w  = (const float*)d_in[12];
  const float* fc2_b  = (const float*)d_in[13];
  const float* fns    = (const float*)d_in[14];
  const float* fnb    = (const float*)d_in[15];
  const float* head_w = (const float*)d_in[16];
  const float* head_b = (const float*)d_in[17];

  char* ws = (char*)d_ws;
  size_t off = 0;
  float* h    = (float*)(ws + off); off += (size_t)MROWS * 768 * 4;   // residual stream (fp32)
  bf16*  y    = (bf16*)(ws + off);  off += (size_t)MROWS * 768 * 2;   // LN output
  bf16*  qkvb = (bf16*)(ws + off);  off += (size_t)MROWS * 2304 * 2;  // qkv
  bf16*  vt   = (bf16*)(ws + off);  off += (size_t)BATCH * HEADS * 96 * NPATCH * 2;
  bf16*  ob   = (bf16*)(ws + off);  off += (size_t)MROWS * 768 * 2;   // attn out
  bf16*  mlp  = (bf16*)(ws + off);  off += (size_t)MROWS * 3072 * 2;  // gelu(fc1)
  bf16*  pcol = (bf16*)(ws + off);  off += (size_t)MROWS * 1536 * 2;  // im2col
  bf16*  pooled = (bf16*)(ws + off);

  const dim3 blk(256);
  const int mT = (MROWS + 127) / 128;

  // Patch embed: h = im2col(x) * conv_w^T + conv_b + pos
  im2col_kernel<<<MROWS, 256, 0, stream>>>(x, pcol);
  gemm_bf16<4><<<dim3(EMBED / 128, mT), blk, 0, stream>>>(
      pcol, conv_w, MROWS, EMBED, 1536, conv_b, nullptr, h, nullptr);

  for (int l = 0; l < DEPTH; ++l) {
    ln_kernel<<<MROWS, 256, 0, stream>>>(h, n1s + l * 768, n1b + l * 768, y);
    gemm_bf16<2><<<dim3(2304 / 128, mT), blk, 0, stream>>>(
        y, qkv_w + (size_t)l * 2304 * 768, MROWS, 2304, 768, nullptr, nullptr,
        nullptr, qkvb);
    vtrans_kernel<<<BATCH * HEADS * 96, 256, 0, stream>>>(qkvb, vt);
    attn_kernel<<<BATCH * HEADS * 25, 128, 0, stream>>>(qkvb, vt, ob);
    gemm_bf16<1><<<dim3(EMBED / 128, mT), blk, 0, stream>>>(
        ob, out_w + (size_t)l * 768 * 768, MROWS, 768, 768, out_b + l * 768, h,
        h, nullptr);
    ln_kernel<<<MROWS, 256, 0, stream>>>(h, n2s + l * 768, n2b + l * 768, y);
    gemm_bf16<3><<<dim3(MLPH / 128, mT), blk, 0, stream>>>(
        y, fc1_w + (size_t)l * 3072 * 768, MROWS, 3072, 768, fc1_b + l * 3072,
        nullptr, nullptr, mlp);
    gemm_bf16<1><<<dim3(EMBED / 128, mT), blk, 0, stream>>>(
        mlp, fc2_w + (size_t)l * 768 * 3072, MROWS, 768, 3072, fc2_b + l * 768,
        h, h, nullptr);
  }

  pool_ln_kernel<<<BATCH, 768, 0, stream>>>(h, fns, fnb, pooled);
  gemm_bf16<0><<<dim3((NCLS + 127) / 128, 1), blk, 0, stream>>>(
      pooled, head_w, BATCH, NCLS, 768, head_b, nullptr, (float*)d_out,
      nullptr);
}
